// GraphEmbeddingNet_21105469293019
// MI455X (gfx1250) — matrix operand order repaired and checked by reading, not compiled, not run on hardware
//
#include <hip/hip_runtime.h>

typedef _Float16 half_t;
typedef __attribute__((ext_vector_type(16))) _Float16 v16h;
typedef __attribute__((ext_vector_type(8)))  _Float16 v8h;
typedef __attribute__((ext_vector_type(8)))  float    v8f;

// Problem constants (match reference)
#define NN   50000
#define EE   800000
#define FF   32
#define DD   32
#define EHH  96
#define GG   128
#define NGG  500
#define LL   3

#define FRAG 512   // halves per WMMA fragment in LDS: 32 lanes x 16 halves

__device__ __forceinline__ float sigf(float x) { return 1.0f / (1.0f + __expf(-x)); }

// Fragment load: two ds_load_b128 (16B each), no per-half packing.
__device__ __forceinline__ v16h ld_frag(const half_t* p) {
    v8h lo = *(const v8h*)p;
    v8h hi = *(const v8h*)(p + 8);
    return __builtin_shufflevector(lo, hi, 0, 1, 2, 3, 4, 5, 6, 7,
                                   8, 9, 10, 11, 12, 13, 14, 15);
}

// ---------------- Encoder: h = X @ enc_W + enc_b  [N,32] (f32 + f16 shadow) ----------------
__global__ void enc_kernel(const float* __restrict__ X, const float* __restrict__ W,
                           const float* __restrict__ b, float* __restrict__ h,
                           half_t* __restrict__ hh) {
    int t = blockIdx.x * blockDim.x + threadIdx.x;
    if (t >= NN * DD) return;
    int n = t >> 5, d = t & 31;
    const float* x = X + n * FF;
    float acc = b[d];
#pragma unroll
    for (int k = 0; k < FF; k++) acc += x[k] * W[k * DD + d];
    h[t] = acc;
    hh[t] = (half_t)acc;
}

// ---------------- Edge message MLP (both directions) via WMMA ----------------
// Per wave iteration: TWO 16-edge tiles (A0, A1 share every B fragment).
// Weights pre-swizzled into per-fragment LDS layout -> B loads = 2x ds_load_b128.
// Node states gathered from the f16 shadow of h (no cvt in the hot loop).
__launch_bounds__(128)
__global__ void edge_msg_kernel(const half_t* __restrict__ hh,
                                const int* __restrict__ from_idx,
                                const int* __restrict__ to_idx,
                                const float* __restrict__ W1f, const float* __restrict__ b1f,
                                const float* __restrict__ W2f, const float* __restrict__ b2f,
                                const float* __restrict__ W1r, const float* __restrict__ b1r,
                                const float* __restrict__ W2r, const float* __restrict__ b2r,
                                float* __restrict__ agg) {
    extern __shared__ char smem[];
    // 60 weight fragments: [dir][0..11]=W1 (ks*6+n), [dir][12..29]=W2 (12+ks*6+n)
    half_t* sW = (half_t*)smem;
    half_t* tilesBase = sW + 60 * FRAG;          // per wave: 4 xcat frags + 6 mid frags
    float*  sB = (float*)(tilesBase + 4 * 10 * FRAG);  // b1f|b1r|b2f|b2r, 96 each
    int*    sIdxAll = (int*)(sB + 4 * 96);       // per wave: 64 ints (2 tiles x from/to)

    const int wave = threadIdx.x >> 5;
    const int lane = threadIdx.x & 31;
    half_t* xhs  = tilesBase + wave * (10 * FRAG);  // [tile][ks] fragments (4)
    half_t* mids = xhs + 4 * FRAG;                  // [tile][ks] fragments (6)
    int*    sIdx = sIdxAll + wave * 64;

    // ---- One-time: swizzle weights (f32 -> f16) into fragment layout ----
    for (int j = threadIdx.x; j < 60 * FRAG; j += blockDim.x) {
        int fragId = j >> 9;      // 0..59
        int slot = j & 511;
        int ln = slot >> 4, i = slot & 15;
        int dir = fragId / 30;
        int g = fragId % 30;
        int ks, n;
        const float* src;
        if (g < 12) { ks = g / 6; n = g % 6; src = dir ? W1r : W1f; }
        else        { g -= 12; ks = g / 6; n = g % 6; src = dir ? W2r : W2f; }
        // B 32x16 layout: lanes 0-15 hold K=kbase+0..7,16..23, lanes 16-31 K=kbase+8..15,24..31
        int k = ks * 32 + ((i >> 3) << 4) + ((ln >> 4) << 3) + (i & 7);
        int col = n * 16 + (ln & 15);
        sW[j] = (half_t)src[k * 96 + col];
    }
    for (int i = threadIdx.x; i < 96; i += blockDim.x) {
        sB[i] = b1f[i]; sB[96 + i] = b1r[i];
        sB[192 + i] = b2f[i]; sB[288 + i] = b2r[i];
    }
    __syncthreads();

    const int nPairs = (EE >> 4) >> 1;   // 25000 pairs of 16-edge tiles
    // Staging constants for this lane (lane == k within a 32-wide k-step)
    const int s_hi = (lane >> 3) & 1;
    const int s_i  = ((lane >> 4) << 3) + (lane & 7);
    const int m  = lane & 15;            // C-fragment column for this lane
    const int hi = lane >> 4;
    const int m0 = hi * 8;               // C-fragment row base
    const v8f vzero = {0.f, 0.f, 0.f, 0.f, 0.f, 0.f, 0.f, 0.f};

    for (int pair = blockIdx.x * 4 + wave; pair < nPairs; pair += gridDim.x * 4) {
        const int e0 = (pair * 2) << 4;
        const int e1 = e0 + 16;
        sIdx[lane]      = (lane < 16) ? from_idx[e0 + lane] : to_idx[e0 + lane - 16];
        sIdx[32 + lane] = (lane < 16) ? from_idx[e1 + lane] : to_idx[e1 + lane - 16];
        asm volatile("s_wait_dscnt 0" ::: "memory");

        // ---- Stage xcat in A-fragment layout: frag[tile][ks], lane holds column k ----
#pragma unroll
        for (int tt = 0; tt < 2; tt++) {
            const int* sI = sIdx + tt * 32;
            half_t* xf = xhs + tt * 2 * FRAG;
#pragma unroll 4
            for (int e = 0; e < 16; e++) {
                int fi = sI[e], ti = sI[16 + e];
                int off = ((s_hi << 4) + e) * 16 + s_i;
                xf[off]        = hh[fi * DD + lane];   // fs -> ks 0 (f16, no cvt)
                xf[FRAG + off] = hh[ti * DD + lane];   // ts -> ks 1
            }
        }
        asm volatile("s_wait_dscnt 0" ::: "memory");

#pragma unroll
        for (int dir = 0; dir < 2; dir++) {
            const half_t* Wd = sW + dir * 30 * FRAG;
            const float*  B1 = sB + dir * 96;
            const float*  B2 = sB + 192 + dir * 96;

            // ---- GEMM1: [16x64] @ [64x96], two tiles share every B fragment ----
            v8f c1[12];
#pragma unroll
            for (int n = 0; n < 12; n++) c1[n] = vzero;
#pragma unroll
            for (int ks = 0; ks < 2; ks++) {
                const int kx = ks ^ dir;   // reverse dir: concat(ts|fs) = swap k-steps
                v16h a0 = ld_frag(xhs + kx * FRAG + lane * 16);
                v16h a1 = ld_frag(xhs + 2 * FRAG + kx * FRAG + lane * 16);
#pragma unroll
                for (int n = 0; n < 6; n++) {
                    v16h bf = ld_frag(Wd + (ks * 6 + n) * FRAG + lane * 16);
                    c1[n]     = __builtin_amdgcn_wmma_f32_16x16x32_f16(
                                    false, a0, false, bf, (short)0, c1[n], false, false);
                    c1[6 + n] = __builtin_amdgcn_wmma_f32_16x16x32_f16(
                                    false, a1, false, bf, (short)0, c1[6 + n], false, false);
                }
            }
            // ---- bias + ReLU -> mids, written directly in A-fragment layout ----
#pragma unroll
            for (int tt = 0; tt < 2; tt++) {
                half_t* md = mids + tt * 3 * FRAG;
#pragma unroll
                for (int n = 0; n < 6; n++) {
                    const int col = n * 16 + m;
                    const int ks2 = col >> 5, kk = col & 31;
                    const int i2 = ((kk >> 4) << 3) + (kk & 7);
                    const int lane2 = (((kk >> 3) & 1) << 4) + m0;
                    const float bias = B1[col];
                    half_t* dst = md + ks2 * FRAG + i2;
#pragma unroll
                    for (int r = 0; r < 8; r++) {
                        float v = c1[tt * 6 + n][r] + bias;
                        v = v > 0.f ? v : 0.f;
                        dst[(lane2 + r) * 16] = (half_t)v;
                    }
                }
            }
            asm volatile("s_wait_dscnt 0" ::: "memory");

            // ---- GEMM2: [16x96] @ [96x96] ----
            v8f c2[12];
#pragma unroll
            for (int n = 0; n < 12; n++) c2[n] = vzero;
#pragma unroll
            for (int ks = 0; ks < 3; ks++) {
                v16h a0 = ld_frag(mids + ks * FRAG + lane * 16);
                v16h a1 = ld_frag(mids + 3 * FRAG + ks * FRAG + lane * 16);
#pragma unroll
                for (int n = 0; n < 6; n++) {
                    v16h bf = ld_frag(Wd + (12 + ks * 6 + n) * FRAG + lane * 16);
                    c2[n]     = __builtin_amdgcn_wmma_f32_16x16x32_f16(
                                    false, a0, false, bf, (short)0, c2[n], false, false);
                    c2[6 + n] = __builtin_amdgcn_wmma_f32_16x16x32_f16(
                                    false, a1, false, bf, (short)0, c2[6 + n], false, false);
                }
            }
            // ---- bias + scatter-add (segment_sum; agg is L2-resident) ----
#pragma unroll
            for (int tt = 0; tt < 2; tt++) {
                const int* dsti = (dir == 0) ? (sIdx + tt * 32 + 16) : (sIdx + tt * 32);
#pragma unroll
                for (int n = 0; n < 6; n++) {
                    const float bias = B2[n * 16 + m];
#pragma unroll
                    for (int r = 0; r < 8; r++) {
                        int node = dsti[m0 + r];
                        atomicAdd(&agg[node * EHH + n * 16 + m], c2[tt * 6 + n][r] + bias);
                    }
                }
            }
        } // dir
    } // pairs
}

// ---------------- GRU update: h' = (1-z)*n + z*h (f32 + f16 shadow) ----------------
__global__ void gru_kernel(const float* __restrict__ agg, const float* __restrict__ hin,
                           const float* __restrict__ Wih, const float* __restrict__ Whh,
                           const float* __restrict__ bih, const float* __restrict__ bhh,
                           float* __restrict__ hout, half_t* __restrict__ hhout) {
    int t = blockIdx.x * blockDim.x + threadIdx.x;
    if (t >= NN * DD) return;
    int n = t >> 5, j = t & 31;
    const float* x  = agg + n * EHH;
    const float* hv = hin + n * DD;
    float gi[3], gh[3];
#pragma unroll
    for (int g = 0; g < 3; g++) {
        int row = g * 32 + j;
        float a = bih[row];
        const float* w = Wih + row * EHH;
#pragma unroll 8
        for (int k = 0; k < EHH; k++) a += x[k] * w[k];
        gi[g] = a;
        float b = bhh[row];
        const float* w2 = Whh + row * DD;
#pragma unroll
        for (int k = 0; k < DD; k++) b += hv[k] * w2[k];
        gh[g] = b;
    }
    float r  = sigf(gi[0] + gh[0]);
    float z  = sigf(gi[1] + gh[1]);
    float nn = tanhf(gi[2] + r * gh[2]);
    float hnew = (1.0f - z) * nn + z * hv[j];
    hout[t] = hnew;
    hhout[t] = (half_t)hnew;
}

// ---------------- Graph aggregator: gated transform + segment_sum ----------------
__global__ void agg_gate_kernel(const float* __restrict__ h, const float* __restrict__ W1,
                                const float* __restrict__ b1, const int* __restrict__ graph_idx,
                                float* __restrict__ gs) {
    int t = blockIdx.x * blockDim.x + threadIdx.x;
    if (t >= NN * GG) return;
    int n = t >> 7, g = t & 127;
    const float* x = h + n * DD;
    float a = b1[g], v = b1[GG + g];
#pragma unroll
    for (int k = 0; k < DD; k++) {
        float xv = x[k];
        a += xv * W1[k * (2 * GG) + g];
        v += xv * W1[k * (2 * GG) + GG + g];
    }
    atomicAdd(&gs[graph_idx[n] * GG + g], sigf(a) * v);
}

// ---------------- Final: out = gs @ agg_W2 + agg_b2  [500,128] ----------------
__global__ void final_kernel(const float* __restrict__ gs, const float* __restrict__ W2,
                             const float* __restrict__ b2, float* __restrict__ out) {
    int t = blockIdx.x * blockDim.x + threadIdx.x;
    if (t >= NGG * GG) return;
    int gi = t >> 7, c = t & 127;
    const float* x = gs + gi * GG;
    float a = b2[c];
#pragma unroll 8
    for (int k = 0; k < GG; k++) a += x[k] * W2[k * GG + c];
    out[t] = a;
}

extern "C" void kernel_launch(void* const* d_in, const int* in_sizes, int n_in,
                              void* d_out, int out_size, void* d_ws, size_t ws_size,
                              hipStream_t stream) {
    const float* X      = (const float*)d_in[0];
    const float* encW   = (const float*)d_in[1];
    const float* encb   = (const float*)d_in[2];
    const float* msgW1  = (const float*)d_in[3];
    const float* msgb1  = (const float*)d_in[4];
    const float* msgW2  = (const float*)d_in[5];
    const float* msgb2  = (const float*)d_in[6];
    const float* rmsgW1 = (const float*)d_in[7];
    const float* rmsgb1 = (const float*)d_in[8];
    const float* rmsgW2 = (const float*)d_in[9];
    const float* rmsgb2 = (const float*)d_in[10];
    const float* gruWih = (const float*)d_in[11];
    const float* gruWhh = (const float*)d_in[12];
    const float* grubih = (const float*)d_in[13];
    const float* grubhh = (const float*)d_in[14];
    const float* aggW1  = (const float*)d_in[15];
    const float* aggb1  = (const float*)d_in[16];
    const float* aggW2  = (const float*)d_in[17];
    const float* aggb2  = (const float*)d_in[18];
    const int*   fromI  = (const int*)d_in[19];
    const int*   toI    = (const int*)d_in[20];
    const int*   gIdx   = (const int*)d_in[21];

    // Workspace layout: h0 | h1 | agg | gs (f32), then hh0 | hh1 (f16)
    float* h0  = (float*)d_ws;
    float* h1  = h0 + (size_t)NN * DD;
    float* agg = h1 + (size_t)NN * DD;
    float* gs  = agg + (size_t)NN * EHH;
    half_t* hh0 = (half_t*)(gs + (size_t)NGG * GG);
    half_t* hh1 = hh0 + (size_t)NN * DD;

    enc_kernel<<<(NN * DD + 255) / 256, 256, 0, stream>>>(X, encW, encb, h0, hh0);

    const size_t shmem = (size_t)(60 * FRAG + 4 * 10 * FRAG) * sizeof(half_t)
                       + (size_t)(4 * 96) * sizeof(float)
                       + (size_t)(4 * 64) * sizeof(int);

    float*  hc = h0;   float*  hn = h1;
    half_t* hhc = hh0; half_t* hhn = hh1;
    for (int l = 0; l < LL; l++) {
        hipMemsetAsync(agg, 0, (size_t)NN * EHH * sizeof(float), stream);
        edge_msg_kernel<<<1024, 128, shmem, stream>>>(
            hhc, fromI, toI,
            msgW1 + (size_t)l * 64 * 96,  msgb1 + (size_t)l * 96,
            msgW2 + (size_t)l * 96 * 96,  msgb2 + (size_t)l * 96,
            rmsgW1 + (size_t)l * 64 * 96, rmsgb1 + (size_t)l * 96,
            rmsgW2 + (size_t)l * 96 * 96, rmsgb2 + (size_t)l * 96,
            agg);
        gru_kernel<<<(NN * DD + 255) / 256, 256, 0, stream>>>(
            agg, hc,
            gruWih + (size_t)l * 96 * EHH, gruWhh + (size_t)l * 96 * DD,
            grubih + (size_t)l * 96,       grubhh + (size_t)l * 96,
            hn, hhn);
        float* tf = hc; hc = hn; hn = tf;
        half_t* th = hhc; hhc = hhn; hhn = th;
    }

    hipMemsetAsync(gs, 0, (size_t)NGG * GG * sizeof(float), stream);
    agg_gate_kernel<<<(NN * GG + 255) / 256, 256, 0, stream>>>(hc, aggW1, aggb1, gIdx, gs);
    final_kernel<<<(NGG * GG + 255) / 256, 256, 0, stream>>>(gs, aggW2, aggb2, (float*)d_out);
}
